// Retention_80212809220149
// MI455X (gfx1250) — matrix operand order repaired and checked by reading, not compiled, not run on hardware
//
#include <hip/hip_runtime.h>

// ---------------------------------------------------------------------------
// Retention on gfx1250: 3 bf16-WMMA GEMMs + decay mask.
//   qkv = x @ W^T + b ; q,k,v split ; scores = (q k^T)/32 * alpha^(i-j) [i>=j]
//   out = scores @ v
// ---------------------------------------------------------------------------

typedef __attribute__((ext_vector_type(16))) __bf16 v16bf;
typedef __attribute__((ext_vector_type(8)))  float  v8f;

#define BATCH   4
#define SEQ     2048
#define DEMB    1024
#define N3D     3072
#define SCALE   0.03125f                 // 1/sqrt(1024)
#define L2ALPHA (-0.014499569695115089f) // log2(0.99)

union FragU { uint4 u[2]; v16bf v; };

__device__ __forceinline__ unsigned short f2bf(float f) {
  unsigned int u = __float_as_uint(f);
  u += 0x7FFFu + ((u >> 16) & 1u);       // round-to-nearest-even
  return (unsigned short)(u >> 16);
}

// ---------------------------------------------------------------------------
// K0: fp32 -> bf16 convert (vectorized x4; all sizes divisible by 4)
// ---------------------------------------------------------------------------
__global__ __launch_bounds__(256)
void cvt_f32_bf16(const float* __restrict__ src, unsigned short* __restrict__ dst, int n4) {
  int i = blockIdx.x * blockDim.x + threadIdx.x;
  if (i >= n4) return;
  float4 f = ((const float4*)src)[i];
  uint2 p;
  p.x = (unsigned)f2bf(f.x) | ((unsigned)f2bf(f.y) << 16);
  p.y = (unsigned)f2bf(f.z) | ((unsigned)f2bf(f.w) << 16);
  ((uint2*)dst)[i] = p;
}

// ---------------------------------------------------------------------------
// K1: qkv = xb @ wb^T + bias.  Wave tile 32x64 (8 accums), block = 8 waves
// covering M=64 x N=256.  Writes q[m][n], k[m][n-1024], vt[b][d][s].
// ---------------------------------------------------------------------------
__global__ __launch_bounds__(256)
void qkv_gemm(const unsigned short* __restrict__ xb, const unsigned short* __restrict__ wb,
              const float* __restrict__ bias,
              unsigned short* __restrict__ qb, unsigned short* __restrict__ kb,
              unsigned short* __restrict__ vtb) {
  const int lane = threadIdx.x & 31;
  const int wave = threadIdx.x >> 5;
  const int lrow = lane & 15;
  const int hi   = lane >> 4;
  const int mBase = blockIdx.x * 64  + (wave & 1) * 32;   // [0,8192)
  const int nBase = blockIdx.y * 256 + (wave >> 1) * 64;  // [0,3072)

  v8f acc[2][4] = {};

  const unsigned short* aRow0 = xb + (size_t)(mBase + lrow) * DEMB + hi * 8;
  const unsigned short* aRow1 = aRow0 + (size_t)16 * DEMB;
  const unsigned short* bRow  = wb + (size_t)(nBase + lrow) * DEMB + hi * 8;

  for (int k = 0; k < DEMB; k += 32) {
    __builtin_prefetch(aRow0 + k + 256, 0, 0);
    FragU a0, a1;
    a0.u[0] = *(const uint4*)(aRow0 + k);
    a0.u[1] = *(const uint4*)(aRow0 + k + 16);
    a1.u[0] = *(const uint4*)(aRow1 + k);
    a1.u[1] = *(const uint4*)(aRow1 + k + 16);
#pragma unroll
    for (int f = 0; f < 4; ++f) {
      const unsigned short* bp = bRow + (size_t)f * 16 * DEMB + k;
      FragU bb;
      bb.u[0] = *(const uint4*)(bp);
      bb.u[1] = *(const uint4*)(bp + 16);
      acc[0][f] = __builtin_amdgcn_wmma_f32_16x16x32_bf16(false, a0.v, false, bb.v,
                                                          (short)0, acc[0][f], false, false);
      acc[1][f] = __builtin_amdgcn_wmma_f32_16x16x32_bf16(false, a1.v, false, bb.v,
                                                          (short)0, acc[1][f], false, false);
    }
  }

#pragma unroll
  for (int f = 0; f < 4; ++f) {
    const int n = nBase + f * 16 + lrow;        // C col = lane%16
    const float bv = bias[n];
#pragma unroll
    for (int g = 0; g < 2; ++g) {
      const int m0 = mBase + g * 16 + hi * 8;   // C row = r + (lane>=16 ? 8 : 0)
#pragma unroll
      for (int r = 0; r < 8; ++r) {
        const unsigned short h = f2bf(acc[g][f][r] + bv);
        const int m = m0 + r;
        if (n < DEMB) {
          qb[(size_t)m * DEMB + n] = h;
        } else if (n < 2 * DEMB) {
          kb[(size_t)m * DEMB + (n - DEMB)] = h;
        } else {                                 // v stored transposed: vt[b][d][s]
          const int b = m >> 11, s = m & (SEQ - 1), d = n - 2 * DEMB;
          vtb[((size_t)b * DEMB + d) * SEQ + s] = h;
        }
      }
    }
  }
}

// ---------------------------------------------------------------------------
// K2: scores[b][i][j] = f2bf( (q.k^T)/32 * alpha^(i-j) , 0 if j>i ).
// Wave tile 32(i) x 64(j); tiles fully above diagonal just store zeros.
// ---------------------------------------------------------------------------
__global__ __launch_bounds__(256)
void scores_gemm(const unsigned short* __restrict__ qb, const unsigned short* __restrict__ kb,
                 unsigned short* __restrict__ sb) {
  const int lane = threadIdx.x & 31;
  const int wave = threadIdx.x >> 5;
  const int lrow = lane & 15;
  const int hi   = lane >> 4;
  const int b    = blockIdx.z;
  const int mBase = blockIdx.x * 64  + (wave & 1) * 32;   // i
  const int nBase = blockIdx.y * 256 + (wave >> 1) * 64;  // j

  unsigned short* sB = sb + (size_t)b * SEQ * SEQ;

  if (nBase > mBase + 31) {                // fully masked tile -> zeros
#pragma unroll
    for (int f = 0; f < 4; ++f) {
      const int j = nBase + f * 16 + lrow;
#pragma unroll
      for (int g = 0; g < 2; ++g) {
        const int i0 = mBase + g * 16 + hi * 8;
#pragma unroll
        for (int r = 0; r < 8; ++r) sB[(size_t)(i0 + r) * SEQ + j] = 0;
      }
    }
    return;
  }

  v8f acc[2][4] = {};
  const unsigned short* aRow0 = qb + ((size_t)b * SEQ + mBase + lrow) * DEMB + hi * 8;
  const unsigned short* aRow1 = aRow0 + (size_t)16 * DEMB;
  const unsigned short* bRow  = kb + ((size_t)b * SEQ + nBase + lrow) * DEMB + hi * 8;

  for (int k = 0; k < DEMB; k += 32) {
    FragU a0, a1;
    a0.u[0] = *(const uint4*)(aRow0 + k);
    a0.u[1] = *(const uint4*)(aRow0 + k + 16);
    a1.u[0] = *(const uint4*)(aRow1 + k);
    a1.u[1] = *(const uint4*)(aRow1 + k + 16);
#pragma unroll
    for (int f = 0; f < 4; ++f) {
      const unsigned short* bp = bRow + (size_t)f * 16 * DEMB + k;
      FragU bb;
      bb.u[0] = *(const uint4*)(bp);
      bb.u[1] = *(const uint4*)(bp + 16);
      acc[0][f] = __builtin_amdgcn_wmma_f32_16x16x32_bf16(false, a0.v, false, bb.v,
                                                          (short)0, acc[0][f], false, false);
      acc[1][f] = __builtin_amdgcn_wmma_f32_16x16x32_bf16(false, a1.v, false, bb.v,
                                                          (short)0, acc[1][f], false, false);
    }
  }

#pragma unroll
  for (int f = 0; f < 4; ++f) {
    const int j = nBase + f * 16 + lrow;
#pragma unroll
    for (int g = 0; g < 2; ++g) {
      const int i0 = mBase + g * 16 + hi * 8;
#pragma unroll
      for (int r = 0; r < 8; ++r) {
        const int i = i0 + r;
        float v = 0.0f;
        if (j <= i)
          v = acc[g][f][r] * SCALE * __builtin_amdgcn_exp2f((float)(i - j) * L2ALPHA);
        sB[(size_t)i * SEQ + j] = f2bf(v);
      }
    }
  }
}

// ---------------------------------------------------------------------------
// K3: out[b][s][d] = scores[b] @ v[b].  B-operand from vt[b][d][s] so the
// K (=sequence) axis is contiguous.  K-loop truncated at causal boundary.
// ---------------------------------------------------------------------------
__global__ __launch_bounds__(256)
void out_gemm(const unsigned short* __restrict__ sb, const unsigned short* __restrict__ vtb,
              float* __restrict__ out) {
  const int lane = threadIdx.x & 31;
  const int wave = threadIdx.x >> 5;
  const int lrow = lane & 15;
  const int hi   = lane >> 4;
  const int b    = blockIdx.z;
  const int mBase = blockIdx.x * 64  + (wave & 1) * 32;   // s
  const int nBase = blockIdx.y * 256 + (wave >> 1) * 64;  // d

  const unsigned short* sB = sb  + (size_t)b * SEQ * SEQ;
  const unsigned short* vB = vtb + (size_t)b * DEMB * SEQ;

  v8f acc[2][4] = {};
  const unsigned short* aRow0 = sB + (size_t)(mBase + lrow) * SEQ + hi * 8;
  const unsigned short* aRow1 = aRow0 + (size_t)16 * SEQ;
  const unsigned short* bRow  = vB + (size_t)(nBase + lrow) * SEQ + hi * 8;

  const int kMax = mBase + 32;             // scores[j > i] == 0
  for (int k = 0; k < kMax; k += 32) {
    __builtin_prefetch(bRow + k + 256, 0, 0);
    FragU a0, a1;
    a0.u[0] = *(const uint4*)(aRow0 + k);
    a0.u[1] = *(const uint4*)(aRow0 + k + 16);
    a1.u[0] = *(const uint4*)(aRow1 + k);
    a1.u[1] = *(const uint4*)(aRow1 + k + 16);
#pragma unroll
    for (int f = 0; f < 4; ++f) {
      const unsigned short* bp = bRow + (size_t)f * 16 * SEQ + k;
      FragU bb;
      bb.u[0] = *(const uint4*)(bp);
      bb.u[1] = *(const uint4*)(bp + 16);
      acc[0][f] = __builtin_amdgcn_wmma_f32_16x16x32_bf16(false, a0.v, false, bb.v,
                                                          (short)0, acc[0][f], false, false);
      acc[1][f] = __builtin_amdgcn_wmma_f32_16x16x32_bf16(false, a1.v, false, bb.v,
                                                          (short)0, acc[1][f], false, false);
    }
  }

#pragma unroll
  for (int f = 0; f < 4; ++f) {
    const int n = nBase + f * 16 + lrow;
#pragma unroll
    for (int g = 0; g < 2; ++g) {
      const int m0 = mBase + g * 16 + hi * 8;
#pragma unroll
      for (int r = 0; r < 8; ++r)
        out[((size_t)b * SEQ + (m0 + r)) * DEMB + n] = acc[g][f][r];
    }
  }
}

// ---------------------------------------------------------------------------
// Workspace layout (bytes):
//   xb  @ 0          : 8192*1024*2  = 16,777,216
//   wb  @ 16777216   : 3072*1024*2  =  6,291,456
//   qb  @ 23068672   : 16,777,216
//   kb  @ 39845888   : 16,777,216
//   vtb @ 56623104   : 16,777,216   (v transposed: [b][d][s])
//   sb  @ 73400320   : 4*2048*2048*2 = 33,554,432
//   total ~ 102 MB
// ---------------------------------------------------------------------------
extern "C" void kernel_launch(void* const* d_in, const int* in_sizes, int n_in,
                              void* d_out, int out_size, void* d_ws, size_t ws_size,
                              hipStream_t stream) {
  (void)in_sizes; (void)n_in; (void)out_size; (void)ws_size;
  const float* x    = (const float*)d_in[0];
  const float* W    = (const float*)d_in[1];
  const float* bias = (const float*)d_in[2];
  float* out = (float*)d_out;

  char* ws = (char*)d_ws;
  unsigned short* xb  = (unsigned short*)(ws);
  unsigned short* wb  = (unsigned short*)(ws + 16777216);
  unsigned short* qb  = (unsigned short*)(ws + 23068672);
  unsigned short* kb  = (unsigned short*)(ws + 39845888);
  unsigned short* vtb = (unsigned short*)(ws + 56623104);
  unsigned short* sb  = (unsigned short*)(ws + 73400320);

  const int nx4 = (BATCH * SEQ * DEMB) / 4;  // 2,097,152
  const int nw4 = (N3D * DEMB) / 4;          //   786,432
  cvt_f32_bf16<<<(nx4 + 255) / 256, 256, 0, stream>>>(x, xb, nx4);
  cvt_f32_bf16<<<(nw4 + 255) / 256, 256, 0, stream>>>(W, wb, nw4);

  qkv_gemm  <<<dim3(128, 12, 1), 256, 0, stream>>>(xb, wb, bias, qb, kb, vtb);
  scores_gemm<<<dim3(32, 8, BATCH), 256, 0, stream>>>(qb, kb, sb);
  out_gemm  <<<dim3(32, 4, BATCH), 256, 0, stream>>>(sb, vtb, out);
}
